// VectorQuantizer_40776419508570
// MI455X (gfx1250) — matrix operand order repaired and checked by reading, not compiled, not run on hardware
//
#include <hip/hip_runtime.h>

typedef __attribute__((ext_vector_type(16))) __bf16       v16bf;
typedef __attribute__((ext_vector_type(8)))  float        v8f;
typedef __attribute__((ext_vector_type(8)))  unsigned int u32x8;
typedef __attribute__((ext_vector_type(4)))  float        f32x4;
typedef __attribute__((ext_vector_type(8)))  float        f32x8;
typedef __attribute__((ext_vector_type(16))) float        f32x16;

#define KE    64
#define DE    64
#define NROW  (64*128*128)            /* 1,048,576 rows */
#define QN    ((size_t)NROW * DE)     /* 67,108,864 quantized elems */
#define NBLK  1024
#define NWAVE 8
#define ITERS (NROW / (NBLK * NWAVE * 16))  /* 8 */
#define ROWSTRIDE ((size_t)NBLK * NWAVE * 16 * DE)  /* X elems advanced per iteration */

// round-to-nearest-even f32 -> bf16, packed pair (lo = first element) — used once for codebook
__device__ __forceinline__ unsigned int pk2bf(float a, float b) {
  unsigned ua = __builtin_bit_cast(unsigned int, a);
  unsigned ub = __builtin_bit_cast(unsigned int, b);
  ua = (ua + 0x7FFFu + ((ua >> 16) & 1u)) >> 16;
  ub = (ub + 0x7FFFu + ((ub >> 16) & 1u)) >> 16;
  return (ua & 0xFFFFu) | (ub << 16);
}

// concat two f32x4 chunk pairs into a 16-wide vector and fptrunc (RNE) to bf16
__device__ __forceinline__ v16bf mk_a(f32x4 a0, f32x4 a1, f32x4 b0, f32x4 b1) {
  f32x8 lo = __builtin_shufflevector(a0, a1, 0, 1, 2, 3, 4, 5, 6, 7);
  f32x8 hi = __builtin_shufflevector(b0, b1, 0, 1, 2, 3, 4, 5, 6, 7);
  f32x16 f = __builtin_shufflevector(lo, hi, 0, 1, 2, 3, 4, 5, 6, 7,
                                             8, 9, 10, 11, 12, 13, 14, 15);
  return __builtin_convertvector(f, v16bf);   // packed v_cvt bf16 converts
}

__global__ __launch_bounds__(256, 1) void vq_main(
    const float* __restrict__ X, const float* __restrict__ E,
    float* __restrict__ Q, float* __restrict__ IDX,
    float* __restrict__ partials)
{
  __shared__ float        Ef[KE][DE];              // 16 KB f32 codebook (gather + loss)
  __shared__ unsigned int Ebf[KE][DE/2];           //  8 KB bf16(-2*E) packed pairs
  __shared__ float        eqs[KE];                 // ||e_k||^2
  __shared__ float        dist[NWAVE][16][DE + 1]; // ~33 KB distance staging, padded
  __shared__ float        wred[NWAVE];

  const int tid  = threadIdx.x;
  const int wave = tid >> 5;
  const int lane = tid & 31;
  const int r15  = lane & 15;
  const int hs   = lane >> 4;

  // ---- stage codebook ----
  for (int i = tid; i < KE * DE; i += 256) ((float*)Ef)[i] = E[i];
  for (int i = tid; i < KE * DE / 2; i += 256) {
    float a = E[2*i], b = E[2*i+1];
    ((unsigned int*)Ebf)[i] = pk2bf(-2.0f * a, -2.0f * b);
  }
  if (tid < KE) {
    float s = 0.f;
    for (int d = 0; d < DE; ++d) { float v = E[tid*DE + d]; s += v * v; }
    eqs[tid] = s;
  }
  __syncthreads();

  // ---- hoist B operands (constant across all row tiles) ----
  // B is (contraction D) x (codes K) per 16-wide code tile kt; step s covers d in [32s,32s+32).
  // Lane = column (code kt*16+r15); lanes 0-15 hold d=32s+0..7,16..23, lanes 16-31 d=32s+8..15,24..31.
  v16bf B[4][2];
  #pragma unroll
  for (int kt = 0; kt < 4; ++kt) {
    const int col = kt*16 + r15;
    #pragma unroll
    for (int s = 0; s < 2; ++s) {
      u32x8 t;
      #pragma unroll
      for (int i = 0; i < 8; ++i) t[i] = Ebf[col][16*s + 4*hs + (i & 3) + 8*(i >> 2)];
      B[kt][s] = __builtin_bit_cast(v16bf, t);
    }
  }

  // hoist ||e_k||^2 for this lane's 4 columns (loop-invariant)
  float eqv[4];
  #pragma unroll
  for (int kt = 0; kt < 4; ++kt) eqv[kt] = eqs[kt*16 + r15];

  // A-operand chunk starts per 16-bit A layout (16x32): lanes 0-15 get K 0-7 / 16-23,
  // lanes 16-31 get K 8-15 / 24-31 (and +32 for the second WMMA step).
  const int s0 = hs*8, s1 = 16 + hs*8, s2 = 32 + hs*8, s3 = 48 + hs*8;
  float lsum = 0.f;

  for (int it = 0; it < ITERS; ++it) {
    const int rowBase = ((it * NBLK + (int)blockIdx.x) * NWAVE + wave) * 16;
    const float* xr = X + (size_t)(rowBase + r15) * DE;

    // prefetch next iteration's row (fixed +ROWSTRIDE stride); covers both 128B lines
    if (it + 1 < ITERS) {
      __builtin_prefetch(xr + ROWSTRIDE + s0, 0, 3);
      __builtin_prefetch(xr + ROWSTRIDE + s2, 0, 3);
    }

    f32x4 x0a = *(const f32x4*)(xr + s0), x0b = *(const f32x4*)(xr + s0 + 4);
    f32x4 x1a = *(const f32x4*)(xr + s1), x1b = *(const f32x4*)(xr + s1 + 4);
    f32x4 x2a = *(const f32x4*)(xr + s2), x2b = *(const f32x4*)(xr + s2 + 4);
    f32x4 x3a = *(const f32x4*)(xr + s3), x3b = *(const f32x4*)(xr + s3 + 4);

    const v16bf A0 = mk_a(x0a, x0b, x1a, x1b);
    const v16bf A1 = mk_a(x2a, x2b, x3a, x3b);

    // dist[m][k] = ||e_k||^2 - 2 * x_m . e_k   (||x||^2 irrelevant for argmin)
    // C seeded with inline 0; ||e_k||^2 (constant per accumulator column) added post-WMMA.
    #pragma unroll
    for (int kt = 0; kt < 4; ++kt) {
      v8f c = {};
      c = __builtin_amdgcn_wmma_f32_16x16x32_bf16(false, A0, false, B[kt][0],
                                                  (short)0, c, false, false);
      c = __builtin_amdgcn_wmma_f32_16x16x32_bf16(false, A1, false, B[kt][1],
                                                  (short)0, c, false, false);
      #pragma unroll
      for (int r = 0; r < 8; ++r) dist[wave][hs*8 + r][kt*16 + r15] = c[r] + eqv[kt];
    }
    __syncthreads();

    // ---- per-row argmin (first-occurrence tie-break, like jnp.argmin) ----
    float bd = 3.4e38f; int bk = 0;
    #pragma unroll 8
    for (int j = 0; j < 32; ++j) {
      const int   kk = hs*32 + j;
      const float dv = dist[wave][r15][kk];
      if (dv < bd) { bd = dv; bk = kk; }
    }
    const float od = __shfl_xor(bd, 16, 32);
    const int   ok = __shfl_xor(bk, 16, 32);
    if (od < bd || (od == bd && ok < bk)) { bd = od; bk = ok; }
    __syncthreads();   // dist reads done before next iteration overwrites

    if (hs == 0) IDX[rowBase + r15] = (float)bk;

    // ---- gather codebook row (straight-through output) + loss partial ----
    const float* er = &Ef[bk][0];
    float*       qr = Q + (size_t)(rowBase + r15) * DE;
    #pragma unroll
    for (int c4 = 0; c4 < 4; ++c4) {
      const int s = (c4 == 0) ? s0 : (c4 == 1) ? s1 : (c4 == 2) ? s2 : s3;
      const f32x4 e0 = *(const f32x4*)(er + s);
      const f32x4 e1 = *(const f32x4*)(er + s + 4);
      *(f32x4*)(qr + s)     = e0;
      *(f32x4*)(qr + s + 4) = e1;
      const f32x4 xa = (c4 == 0) ? x0a : (c4 == 1) ? x1a : (c4 == 2) ? x2a : x3a;
      const f32x4 xb = (c4 == 0) ? x0b : (c4 == 1) ? x1b : (c4 == 2) ? x2b : x3b;
      f32x4 d0 = e0 - xa, d1 = e1 - xb;
      lsum += d0.x*d0.x + d0.y*d0.y + d0.z*d0.z + d0.w*d0.w
            + d1.x*d1.x + d1.y*d1.y + d1.z*d1.z + d1.w*d1.w;
    }
  }

  // ---- deterministic block reduction of loss partials ----
  #pragma unroll
  for (int off = 16; off > 0; off >>= 1) lsum += __shfl_xor(lsum, off, 32);
  if (lane == 0) wred[wave] = lsum;
  __syncthreads();
  if (tid == 0) {
    float s = 0.f;
    #pragma unroll
    for (int w = 0; w < NWAVE; ++w) s += wred[w];
    partials[blockIdx.x] = s;
  }
}

__global__ __launch_bounds__(256, 1) void vq_finalize(
    const float* __restrict__ partials, float* __restrict__ out_loss)
{
  __shared__ float red[256];
  float s = 0.f;
  for (int i = threadIdx.x; i < NBLK; i += 256) s += partials[i];
  red[threadIdx.x] = s;
  __syncthreads();
  for (int o = 128; o > 0; o >>= 1) {
    if (threadIdx.x < o) red[threadIdx.x] += red[threadIdx.x + o];
    __syncthreads();
  }
  // q_latent + commitment*e_latent = (1 + 0.25) * mean((q - x)^2)
  if (threadIdx.x == 0) out_loss[0] = 1.25f * red[0] / (float)QN;
}

extern "C" void kernel_launch(void* const* d_in, const int* in_sizes, int n_in,
                              void* d_out, int out_size, void* d_ws, size_t ws_size,
                              hipStream_t stream) {
  const float* X = (const float*)d_in[0];   // inputs  [64,128,128,64] f32
  const float* E = (const float*)d_in[1];   // codebook [64,64] f32
  float* out      = (float*)d_out;
  float* Q        = out;                    // quantized_st (67,108,864 f32)
  float* loss     = out + QN;               // vq_loss scalar
  float* IDX      = out + QN + 1;           // encoding indices (as float values)
  float* partials = (float*)d_ws;           // NBLK floats of scratch

  vq_main<<<NBLK, 256, 0, stream>>>(X, E, Q, IDX, partials);
  vq_finalize<<<1, 256, 0, stream>>>(partials, loss);
}
